// MultiHeadSelfAttention_51694226375382
// MI455X (gfx1250) — matrix operand order repaired and checked by reading, not compiled, not run on hardware
//
#include <hip/hip_runtime.h>
#include <hip/hip_bf16.h>

typedef __bf16 bf16;
typedef bf16 bf16x8  __attribute__((ext_vector_type(8)));
typedef bf16 bf16x16 __attribute__((ext_vector_type(16)));
typedef float f32x8  __attribute__((ext_vector_type(8)));

constexpr int kT    = 49;   // tokens per window
constexpr int kTP   = 64;   // padded tokens (4 x 16 WMMA tiles)
constexpr int kF    = 192;  // input / output channels
constexpr int kNH   = 6;    // heads
constexpr int kHD   = 32;   // head dim (attn and hidden)
constexpr int kPERH = 96;   // qkv channels per head (32 Q + 32 K + 32 V)

// ---------------------------------------------------------------------------
// WMMA fragment helpers (wave32; A/B use contiguous k = 16h+s in both, C/D per CDNA5 ISA 7.12.2)
// ---------------------------------------------------------------------------

// A matrix 16x32 bf16, row-major source with leading dim ld (ld % 16 == 0).
// lane L: row = L&15; halves 0..15 -> k = (L>>4)*16 .. +15 (32B contiguous).
__device__ inline bf16x16 load_a(const bf16* base, int ld, int lane) {
  const bf16* p = base + (lane & 15) * ld + ((lane >> 4) << 4);
  return *(const bf16x16*)(p);
}

// B matrix 32x16 bf16. Source is "column-major as rows": row n of `base`
// (leading dim ld, ld % 16 == 0) holds B[k][n] contiguous in k.
// lane L: col = L&15; halves 0..15 -> k = (L>>4)*16 .. +15 (32B contiguous).
__device__ inline bf16x16 load_b(const bf16* base, int ld, int lane) {
  const bf16* p = base + (lane & 15) * ld + ((lane >> 4) << 4);
  return *(const bf16x16*)(p);
}

// C/D 16x16 f32: VGPR r <-> row (lane>>4)*8 + r, col = lane&15.
__device__ inline void store_c(float* base, int ld, int lane, f32x8 c) {
  float* p = base + (((lane >> 4) << 3)) * ld + (lane & 15);
#pragma unroll
  for (int r = 0; r < 8; ++r) p[r * ld] = c[r];
}

__device__ inline f32x8 wmma_bf16(bf16x16 a, bf16x16 b, f32x8 c) {
  return __builtin_amdgcn_wmma_f32_16x16x32_bf16(false, a, false, b,
                                                 (short)0, c, false, false);
}

// ---------------------------------------------------------------------------
// LDS layout (~130 KB; 2 workgroups per WGP under the 320 KB limit)
// ---------------------------------------------------------------------------
struct __align__(64) SM {
  union {
    bf16 X[kTP * kF];                // stage 1: bf16 X, zero-padded rows
    bf16 O[kTP * kF];                // stage 3 output / stage 4 A operand
  } xo;                              // 24576 B
  bf16 Q[kNH * kTP * kHD];           // 24576 B (post-LN)
  bf16 K[kNH * kTP * kHD];           // 24576 B (post-LN)
  bf16 Vt[kNH * kHD * kTP];          // 24576 B (V transposed: [h][d][tok])
  union {
    float qkv[kTP * kPERH];          // per-head fp32 GEMM1 staging
    struct {
      float S[kTP * kTP];            // logits fp32
      bf16  A[kTP * kTP];            // softmax probs bf16 (pad cols zeroed)
    } att;
  } u;                               // 24576 B
  int relidx[kT * kT];               // 9604 B
};

// ---------------------------------------------------------------------------
// fp32 -> bf16 weight conversion into workspace (runs once per launch)
// ---------------------------------------------------------------------------
__global__ void cvt_weights(const float* __restrict__ wqkv,
                            const float* __restrict__ wout,
                            bf16* __restrict__ wq, bf16* __restrict__ wo) {
  int i = blockIdx.x * 256 + threadIdx.x;
  if (i < 576 * 192) wq[i] = (bf16)wqkv[i];
  if (i < 192 * 192) wo[i] = (bf16)wout[i];
}

// ---------------------------------------------------------------------------
// Fused windowed MHSA: one workgroup (8 wave32) per window.
// __launch_bounds__(256, 4): ask for >=4 waves per SIMD (2 co-resident
// blocks per WGP) -> cap 256 VGPRs/wave, no s_set_vgpr_msb addressing, so a
// second block's WMMA phases can overlap this block's LN/softmax phases.
// ---------------------------------------------------------------------------
__global__ __launch_bounds__(256, 4) void
fused_window_attn(const float* __restrict__ X,
                  const bf16*  __restrict__ Wq,   // [576][192] bf16
                  const bf16*  __restrict__ Wo,   // [192][192] bf16
                  const float* __restrict__ qg, const float* __restrict__ qb,
                  const float* __restrict__ kg, const float* __restrict__ kb,
                  const float* __restrict__ rtab,  // [169][6]
                  const int*   __restrict__ ridx,  // [49][49]
                  const float* __restrict__ bout,  // [192]
                  float* __restrict__ out) {
  __shared__ SM sm;
  const int w    = blockIdx.x;
  const int tid  = threadIdx.x;
  const int wave = tid >> 5;
  const int lane = tid & 31;
  const float* Xw = X + (size_t)w * kT * kF;

  // ---- stage 0: X -> LDS bf16 (zero-pad rows 49..63), rel_idx -> LDS ----
  for (int i = tid; i < kT * kF; i += 256) sm.xo.X[i] = (bf16)Xw[i];
  for (int i = kT * kF + tid; i < kTP * kF; i += 256) sm.xo.X[i] = (bf16)0.0f;
  for (int i = tid; i < kT * kT; i += 256) sm.relidx[i] = ridx[i];
  __syncthreads();

  // ---- stage 1+2: per-head QKV GEMM (64x96 = X[64x192] * WqT) + LN ----
  for (int h = 0; h < kNH; ++h) {
#pragma unroll
    for (int i = 0; i < 3; ++i) {             // 24 tiles / 8 waves
      int t  = wave * 3 + i;
      int mt = t / 6, nt = t % 6;
      f32x8 acc = {};
#pragma unroll
      for (int kk = 0; kk < 6; ++kk) {        // K = 192
        bf16x16 a = load_a(&sm.xo.X[mt * 16 * kF + kk * 32], kF, lane);
        bf16x16 b = load_b(&Wq[(size_t)(h * kPERH + nt * 16) * kF + kk * 32],
                           kF, lane);
        acc = wmma_bf16(a, b, acc);
      }
      store_c(&sm.u.qkv[mt * 16 * kPERH + nt * 16], kPERH, lane, acc);
    }
    __syncthreads();

    {   // LN epilogue: 4 lanes per token row, all 256 threads active.
        // Cross-lane mean/var via shfl_xor within the aligned 4-lane group.
      const int tok = tid >> 2;               // 0..63
      const int j   = tid & 3;                // handles dims j*8 .. j*8+7
      float* row = &sm.u.qkv[tok * kPERH];

      // ---- Q layernorm ----
      float s = 0.f;
#pragma unroll
      for (int dd = 0; dd < 8; ++dd) s += row[j * 8 + dd];
      s += __shfl_xor(s, 1); s += __shfl_xor(s, 2);
      float mu = s * (1.0f / kHD);
      float v = 0.f;
#pragma unroll
      for (int dd = 0; dd < 8; ++dd) {
        float c = row[j * 8 + dd] - mu; v += c * c;
      }
      v += __shfl_xor(v, 1); v += __shfl_xor(v, 2);
      float rs = rsqrtf(v * (1.0f / kHD) + 1e-5f);
#pragma unroll
      for (int dd = 0; dd < 8; ++dd) {
        int d = j * 8 + dd;
        sm.Q[h * kTP * kHD + tok * kHD + d] =
            (bf16)((row[d] - mu) * rs * qg[d] + qb[d]);
      }

      // ---- K layernorm ----
      s = 0.f;
#pragma unroll
      for (int dd = 0; dd < 8; ++dd) s += row[kHD + j * 8 + dd];
      s += __shfl_xor(s, 1); s += __shfl_xor(s, 2);
      mu = s * (1.0f / kHD);
      v = 0.f;
#pragma unroll
      for (int dd = 0; dd < 8; ++dd) {
        float c = row[kHD + j * 8 + dd] - mu; v += c * c;
      }
      v += __shfl_xor(v, 1); v += __shfl_xor(v, 2);
      rs = rsqrtf(v * (1.0f / kHD) + 1e-5f);
#pragma unroll
      for (int dd = 0; dd < 8; ++dd) {
        int d = j * 8 + dd;
        sm.K[h * kTP * kHD + tok * kHD + d] =
            (bf16)((row[kHD + d] - mu) * rs * kg[d] + kb[d]);
      }

      // ---- V: store transposed [d][tok] so A*V B-operand is k-contiguous ----
#pragma unroll
      for (int dd = 0; dd < 8; ++dd) {
        int d = j * 8 + dd;
        sm.Vt[h * kHD * kTP + d * kTP + tok] = (bf16)row[2 * kHD + d];
      }
    }
    __syncthreads();
  }

  // ---- stage 3: per-head attention (S = QK^T, softmax+bias, O = A V) ----
  for (int h = 0; h < kNH; ++h) {
#pragma unroll
    for (int i = 0; i < 2; ++i) {             // 16 S-tiles / 8 waves
      int t  = wave * 2 + i;
      int mt = t >> 2, nt = t & 3;
      f32x8 acc = {};
      bf16x16 a = load_a(&sm.Q[h * kTP * kHD + mt * 16 * kHD], kHD, lane);
      bf16x16 b = load_b(&sm.K[h * kTP * kHD + nt * 16 * kHD], kHD, lane);
      acc = wmma_bf16(a, b, acc);             // K = head dim = 32, one step
      store_c(&sm.u.att.S[mt * 16 * kTP + nt * 16], kTP, lane, acc);
    }
    __syncthreads();

    {   // softmax: 4 lanes per query row, each covering 16 keys
      const int q = tid >> 2;
      const int j = tid & 3;
      if (q < kT) {
        const float scale = 0.17677669529663687f;  // 1/sqrt(32)
        const int k0   = j * 16;
        const int kend = (k0 + 16 < kT) ? (k0 + 16) : kT;
        float m = -3.4e38f;
        for (int k = k0; k < kend; ++k) {
          float sv = sm.u.att.S[q * kTP + k] * scale +
                     rtab[sm.relidx[q * kT + k] * kNH + h];
          sm.u.att.S[q * kTP + k] = sv;
          m = fmaxf(m, sv);
        }
        m = fmaxf(m, __shfl_xor(m, 1));
        m = fmaxf(m, __shfl_xor(m, 2));
        float sum = 0.f;
        for (int k = k0; k < kend; ++k) {
          float e = __expf(sm.u.att.S[q * kTP + k] - m);
          sm.u.att.S[q * kTP + k] = e;
          sum += e;
        }
        sum += __shfl_xor(sum, 1);
        sum += __shfl_xor(sum, 2);
        float inv = 1.0f / sum;
#pragma unroll
        for (int kk = 0; kk < 16; ++kk) {
          int k = k0 + kk;
          sm.u.att.A[q * kTP + k] =
              (k < kT) ? (bf16)(sm.u.att.S[q * kTP + k] * inv) : (bf16)0.0f;
        }
      }
    }
    __syncthreads();

    {                                         // O = A[64x64] * V[64x32], 8 tiles
      int mt = wave >> 1, nt = wave & 1;
      f32x8 acc = {};
#pragma unroll
      for (int kk = 0; kk < 2; ++kk) {
        bf16x16 a = load_a(&sm.u.att.A[mt * 16 * kTP + kk * 32], kTP, lane);
        bf16x16 b = load_b(&sm.Vt[h * kHD * kTP + nt * 16 * kTP + kk * 32],
                           kTP, lane);
        acc = wmma_bf16(a, b, acc);
      }
      int col  = lane & 15;
      int row0 = mt * 16 + ((lane >> 4) << 3);
#pragma unroll
      for (int r = 0; r < 8; ++r)
        sm.xo.O[(row0 + r) * kF + h * kHD + nt * 16 + col] = (bf16)acc[r];
    }
    __syncthreads();
  }

  // ---- stage 4: out = O[64x192] * Wout^T + b_out, store rows < 49 ----
#pragma unroll
  for (int i = 0; i < 6; ++i) {               // 48 tiles / 8 waves
    int t  = wave * 6 + i;
    int mt = t / 12, nt = t % 12;
    f32x8 acc = {};
#pragma unroll
    for (int kk = 0; kk < 6; ++kk) {          // K = 192
      bf16x16 a = load_a(&sm.xo.O[mt * 16 * kF + kk * 32], kF, lane);
      bf16x16 b = load_b(&Wo[(size_t)(nt * 16) * kF + kk * 32], kF, lane);
      acc = wmma_bf16(a, b, acc);
    }
    int col  = lane & 15;
    int row0 = mt * 16 + ((lane >> 4) << 3);
#pragma unroll
    for (int r = 0; r < 8; ++r) {
      int row = row0 + r;
      if (row < kT)
        out[(size_t)w * kT * kF + row * kF + nt * 16 + col] =
            acc[r] + bout[nt * 16 + col];
    }
  }
}

// ---------------------------------------------------------------------------
extern "C" void kernel_launch(void* const* d_in, const int* in_sizes, int n_in,
                              void* d_out, int out_size, void* d_ws,
                              size_t ws_size, hipStream_t stream) {
  const float* X    = (const float*)d_in[0];
  const float* Wqkv = (const float*)d_in[1];
  const float* qg   = (const float*)d_in[2];
  const float* qb   = (const float*)d_in[3];
  const float* kg   = (const float*)d_in[4];
  const float* kb   = (const float*)d_in[5];
  const float* rtab = (const float*)d_in[6];
  const int*   ridx = (const int*)d_in[7];
  const float* Wout = (const float*)d_in[8];
  const float* bout = (const float*)d_in[9];
  float* out = (float*)d_out;

  bf16* wq = (bf16*)d_ws;             // 576*192 bf16
  bf16* wo = wq + 576 * 192;          // 192*192 bf16 (total ~295 KB of ws)

  const int nwin = in_sizes[0] / (kT * kF);   // 8192

  cvt_weights<<<(576 * 192 + 255) / 256, 256, 0, stream>>>(Wqkv, Wout, wq, wo);
  fused_window_attn<<<nwin, 256, 0, stream>>>(X, wq, wo, qg, qb, kg, kb,
                                              rtab, ridx, bout, out);
}
